// HRMModel_4105988735048
// MI455X (gfx1250) — compile-verified, matrix operand order checked by reading, not run on hardware
//
#include <hip/hip_runtime.h>
#include <hip/hip_bf16.h>
#include <cstdint>

// Model dims
#define Bb 8
#define Ss 256
#define Vv 32000
#define Hh 1024

typedef __attribute__((ext_vector_type(16))) __bf16 v16bf;
typedef __attribute__((ext_vector_type(8)))  float  v8f;

union FragU { uint4 u4[2]; v16bf v; };

__device__ __forceinline__ v8f wmma_bf16(v16bf a, v16bf b, v8f c) {
  // D(16x16,f32) = A(16x32,bf16) x B(32x16,bf16) + C
  return __builtin_amdgcn_wmma_f32_16x16x32_bf16(false, a, false, b, (short)0, c, false, false);
}

// A fragment from LDS-staged bf16 activations [9][K]; rows 0..7 = data, row 8 = zeros.
// ISA 16-bit A layout: lane<16 -> M=lane, K octets {0..7,16..23}; lane>=16 -> M=lane-16, K octets {8..15,24..31}.
// Lanes with M>=8 (padding rows) read the shared zero row -> no divergence, EXEC stays all-ones.
__device__ __forceinline__ const __bf16* afrag_base(const __bf16* sm, int K, int lane) {
  int M = lane & 15, sel = (lane >> 4) & 1;
  int row = (M < 8) ? M : 8;
  return sm + row * K + sel * 8;
}

__device__ __forceinline__ v16bf afrag_ld(const __bf16* base, int kb) {
  FragU f;
  f.u4[0] = *reinterpret_cast<const uint4*>(base + kb * 32);        // K octet 0
  f.u4[1] = *reinterpret_cast<const uint4*>(base + kb * 32 + 16);   // K octet +16
  return f.v;
}

// B fragment: pre-packed 512 bf16 per fragment, lane-major (lane*16 contiguous).
__device__ __forceinline__ v16bf bfrag_ld(const __bf16* Bt, int frag, int lane) {
  const uint4* p = reinterpret_cast<const uint4*>(Bt + (size_t)frag * 512 + lane * 16);
  FragU f; f.u4[0] = p[0]; f.u4[1] = p[1];
  return f.v;
}

// Stage [8][K] f32 activations (optionally concat of two 1024-wide sources) into LDS as
// bf16 [9][K], with row 8 zero-filled (padding row for WMMA M=8..15 lanes).
__device__ __forceinline__ void stage_a(const float* A0, const float* A1, int a1stride,
                                        int K, __bf16* sm, int tid, int nthreads) {
  int total = 9 * K;
  for (int i = tid; i < total; i += nthreads) {
    int r = i / K, c = i - r * K;
    float v = 0.0f;
    if (r < 8) {
      if (A1) v = (c < 1024) ? A0[r * 1024 + c] : A1[r * a1stride + (c - 1024)];
      else    v = A0[(size_t)r * K + c];
    }
    sm[i] = (__bf16)v;
  }
  __syncthreads();
}

// ---------------- weight repack: f32 [N][K] row-major -> bf16 WMMA B-fragments ----------------
__global__ void k_pack(const float* __restrict__ W, __bf16* __restrict__ out, int N, int K) {
  long long total = (long long)N * K;
  int KB = K >> 5;
  for (long long o = (long long)blockIdx.x * blockDim.x + threadIdx.x; o < total;
       o += (long long)gridDim.x * blockDim.x) {
    int i    = (int)(o & 15);
    int lane = (int)((o >> 4) & 31);
    long long frag = o >> 9;
    int kb = (int)(frag % KB);
    int nt = (int)(frag / KB);
    int n  = nt * 16 + (lane & 15);
    int k  = kb * 32 + ((lane >> 4) & 1) * 8 + (i < 8 ? i : i + 8);
    out[o] = (__bf16)W[(size_t)n * K + k];
  }
}

// ---------------- embedding gather ----------------
__global__ void k_embed(const int* __restrict__ ids, const float* __restrict__ emb,
                        const float* __restrict__ pos, float* __restrict__ e, int t) {
  int i = blockIdx.x * blockDim.x + threadIdx.x;
  if (i < Bb * Hh) {
    int b = i >> 10, d = i & 1023;
    int id = ids[b * Ss + t];
    e[i] = emb[(size_t)id * Hh + d] + pos[(size_t)t * Hh + d];
  }
}

// ---------------- generic linear: Out = epi(A @ W^T + bias [+resid]) ----------------
// epi: 0 = none, 1 = gelu(tanh approx), 2 = add resid
template <int K>
__global__ void k_linear(const float* __restrict__ A0, const float* __restrict__ A1, int a1stride,
                         const __bf16* __restrict__ Bp, const float* __restrict__ bias,
                         const float* __restrict__ resid, float* __restrict__ Out, int N, int epi) {
  extern __shared__ __bf16 sm[];
  int tid = threadIdx.x, lane = tid & 31, wave = tid >> 5;
  stage_a(A0, A1, a1stride, K, sm, tid, blockDim.x);
  constexpr int KB = K >> 5;
  int nt = blockIdx.x * 4 + wave;
  const __bf16* Bt = Bp + (size_t)nt * KB * 512;
  const __bf16* Ab = afrag_base(sm, K, lane);
  v8f acc = {};
#pragma unroll 4
  for (int kb = 0; kb < KB; ++kb) {
    __builtin_prefetch(Bt + (size_t)(kb + 2) * 512 + lane * 16, 0, 1);
    acc = wmma_bf16(afrag_ld(Ab, kb), bfrag_ld(Bt, kb, lane), acc);
  }
  if (lane < 16) {
    int col = nt * 16 + lane;
    float bv = bias ? bias[col] : 0.0f;
    for (int j = 0; j < 8; ++j) {
      float v = acc[j] + bv;
      if (epi == 1) {
        float u = 0.7978845608028654f * (v + 0.044715f * v * v * v);
        v = 0.5f * v * (1.0f + tanhf(u));
      } else if (epi == 2) {
        v += resid[(size_t)j * N + col];
      }
      Out[(size_t)j * N + col] = v;
    }
  }
}

// ---------------- GRU cell: out = (1-z)*n + z*h ----------------
__global__ void k_gru(const float* __restrict__ x, const float* __restrict__ h,
                      const float* __restrict__ bih, const float* __restrict__ bhh,
                      const __bf16* __restrict__ Wih_p, const __bf16* __restrict__ Whh_p,
                      float* __restrict__ out) {
  extern __shared__ __bf16 sm[];
  __bf16* smx = sm;                 // [9][1024]
  __bf16* smh = sm + 9 * 1024;      // [9][1024]
  int tid = threadIdx.x, lane = tid & 31, wave = tid >> 5;
  for (int i = tid; i < 2 * 9 * 1024; i += blockDim.x) {
    int half = (i < 9216) ? 0 : 1;
    int j = i - half * 9216;
    float v = 0.0f;
    if (j < 8192) v = half ? h[j] : x[j];
    sm[i] = (__bf16)v;
  }
  __syncthreads();

  int nt = blockIdx.x * 4 + wave;            // 0..63 (output column tile)
  const __bf16* Abx = afrag_base(smx, 1024, lane);
  const __bf16* Abh = afrag_base(smh, 1024, lane);
  v8f aX[3] = {{}, {}, {}};                  // r, z, n gates from x@Wih^T
  v8f aH[3] = {{}, {}, {}};                  // r, z, n gates from h@Whh^T
#pragma unroll 2
  for (int kb = 0; kb < 32; ++kb) {
    v16bf ax = afrag_ld(Abx, kb);
    v16bf ah = afrag_ld(Abh, kb);
#pragma unroll
    for (int g = 0; g < 3; ++g) {
      int frag = (nt + 64 * g) * 32 + kb;
      __builtin_prefetch(Wih_p + (size_t)(frag + 2) * 512 + lane * 16, 0, 1);
      __builtin_prefetch(Whh_p + (size_t)(frag + 2) * 512 + lane * 16, 0, 1);
      aX[g] = wmma_bf16(ax, bfrag_ld(Wih_p, frag, lane), aX[g]);
      aH[g] = wmma_bf16(ah, bfrag_ld(Whh_p, frag, lane), aH[g]);
    }
  }
  if (lane < 16) {
    int c = nt * 16 + lane;
    float bxr = bih[c], bxz = bih[c + 1024], bxn = bih[c + 2048];
    float bhr = bhh[c], bhz = bhh[c + 1024], bhn = bhh[c + 2048];
    for (int j = 0; j < 8; ++j) {
      float xr = aX[0][j] + bxr, hr = aH[0][j] + bhr;
      float xz = aX[1][j] + bxz, hz = aH[1][j] + bhz;
      float xn = aX[2][j] + bxn, hn = aH[2][j] + bhn;
      float r = 1.0f / (1.0f + expf(-(xr + hr)));
      float z = 1.0f / (1.0f + expf(-(xz + hz)));
      float n = tanhf(xn + r * hn);
      float ho = h[(size_t)j * 1024 + c];
      out[(size_t)j * 1024 + c] = (1.0f - z) * n + z * ho;
    }
  }
}

// ---------------- history slot write + K/V cache update for that slot ----------------
__global__ void k_kv_hist(const float* __restrict__ hout, float* __restrict__ hist,
                          float* __restrict__ Kc, float* __restrict__ Vc,
                          const __bf16* __restrict__ Wk_p, const __bf16* __restrict__ Wv_p, int hs) {
  if (blockIdx.x == 32) {  // copy hist[:,hs] = hout
    for (int i = threadIdx.x; i < 8192; i += blockDim.x)
      hist[(size_t)(i >> 10) * 4096 + (size_t)hs * 1024 + (i & 1023)] = hout[i];
    return;
  }
  extern __shared__ __bf16 sm[];
  int tid = threadIdx.x, lane = tid & 31, wave = tid >> 5;
  stage_a(hout, nullptr, 0, 1024, sm, tid, blockDim.x);
  int tile = blockIdx.x * 4 + wave;          // 0..127
  const __bf16* Wp = (tile < 64) ? Wk_p : Wv_p;
  float* Dst = (tile < 64) ? Kc : Vc;
  int nt = tile & 63;
  const __bf16* Bt = Wp + (size_t)nt * 32 * 512;
  const __bf16* Ab = afrag_base(sm, 1024, lane);
  v8f acc = {};
#pragma unroll 4
  for (int kb = 0; kb < 32; ++kb) {
    __builtin_prefetch(Bt + (size_t)(kb + 2) * 512 + lane * 16, 0, 1);
    acc = wmma_bf16(afrag_ld(Ab, kb), bfrag_ld(Bt, kb, lane), acc);
  }
  if (lane < 16) {
    int col = nt * 16 + lane;
    for (int j = 0; j < 8; ++j)
      Dst[(size_t)j * 4096 + (size_t)hs * 1024 + col] = acc[j];
  }
}

// ---------------- attention: one block per head (8 waves) ----------------
__global__ void k_attn(const float* __restrict__ inp, const float* __restrict__ Kc,
                       const float* __restrict__ Vc, const __bf16* __restrict__ Wq_p,
                       float* __restrict__ ctx, int hs) {
  __shared__ float qS[8][128];
  __shared__ float attnS[8][4];
  extern __shared__ __bf16 sm[];
  int tid = threadIdx.x, lane = tid & 31, wave = tid >> 5, head = blockIdx.x;
  stage_a(inp, nullptr, 0, 1024, sm, tid, blockDim.x);

  int nt = head * 8 + wave;                  // q column tile in [0,64)
  const __bf16* Bt = Wq_p + (size_t)nt * 32 * 512;
  const __bf16* Ab = afrag_base(sm, 1024, lane);
  v8f acc = {};
#pragma unroll 4
  for (int kb = 0; kb < 32; ++kb) {
    __builtin_prefetch(Bt + (size_t)(kb + 2) * 512 + lane * 16, 0, 1);
    acc = wmma_bf16(afrag_ld(Ab, kb), bfrag_ld(Bt, kb, lane), acc);
  }
  const float scale = 0.08838834764831845f;  // 1/sqrt(128)
  if (lane < 16) {
    int dl = wave * 16 + lane;
    for (int j = 0; j < 8; ++j) qS[j][dl] = acc[j] * scale;
  }
  __syncthreads();

  if (tid < 32) {  // deterministic scores + softmax: lane = b*4 + tt
    int b = tid >> 2, tt = tid & 3;
    float s = -1.0e9f;
    if (tt <= hs) {
      s = 0.0f;
      const float* kp = Kc + (size_t)b * 4096 + (size_t)tt * 1024 + head * 128;
      for (int d = 0; d < 128; ++d) s += qS[b][d] * kp[d];
    }
    float m = fmaxf(s, __shfl_xor(s, 1, 4));
    m = fmaxf(m, __shfl_xor(m, 2, 4));
    float ex = expf(s - m);
    float sum = ex + __shfl_xor(ex, 1, 4);
    sum += __shfl_xor(sum, 2, 4);
    attnS[b][tt] = ex / sum;
  }
  __syncthreads();

  if (lane < 16) {
    int d = head * 128 + wave * 16 + lane;
    for (int j = 0; j < 8; ++j) {
      float c = 0.0f;
      const float* vp = Vc + (size_t)j * 4096 + d;
      for (int tt = 0; tt < 4; ++tt) c += attnS[j][tt] * vp[(size_t)tt * 1024];
      ctx[(size_t)j * 1024 + d] = c;
    }
  }
}

// ---------------- layernorm ----------------
__global__ void k_ln(const float* __restrict__ xin, const float* __restrict__ g,
                     const float* __restrict__ bb, float* __restrict__ y) {
  int tid = threadIdx.x, row = tid >> 5, lane = tid & 31;
  const float* xr = xin + (size_t)row * 1024;
  float s = 0.0f;
  for (int c = lane; c < 1024; c += 32) s += xr[c];
  for (int o = 16; o; o >>= 1) s += __shfl_xor(s, o, 32);
  float mu = s * (1.0f / 1024.0f);
  float vv = 0.0f;
  for (int c = lane; c < 1024; c += 32) { float d = xr[c] - mu; vv += d * d; }
  for (int o = 16; o; o >>= 1) vv += __shfl_xor(vv, o, 32);
  float rstd = rsqrtf(vv * (1.0f / 1024.0f) + 1e-5f);
  for (int c = lane; c < 1024; c += 32)
    y[(size_t)row * 1024 + c] = (xr[c] - mu) * rstd * g[c] + bb[c];
}

// ---------------- vocab projection: out[b][t][:] = y @ Wout^T ----------------
__global__ void k_logits(const float* __restrict__ y, const __bf16* __restrict__ Wout_p,
                         float* __restrict__ out, int t) {
  extern __shared__ __bf16 sm[];
  int tid = threadIdx.x, lane = tid & 31, wave = tid >> 5;
  stage_a(y, nullptr, 0, 1024, sm, tid, blockDim.x);
  int nt = blockIdx.x * 4 + wave;            // 0..1999
  const __bf16* Bt = Wout_p + (size_t)nt * 32 * 512;
  const __bf16* Ab = afrag_base(sm, 1024, lane);
  v8f acc = {};
#pragma unroll 4
  for (int kb = 0; kb < 32; ++kb) {
    __builtin_prefetch(Bt + (size_t)(kb + 2) * 512 + lane * 16, 0, 1);
    acc = wmma_bf16(afrag_ld(Ab, kb), bfrag_ld(Bt, kb, lane), acc);
  }
  if (lane < 16) {
    int col = nt * 16 + lane;
    for (int j = 0; j < 8; ++j)
      out[(size_t)j * ((size_t)Ss * Vv) + (size_t)t * Vv + col] = acc[j];
  }
}

extern "C" void kernel_launch(void* const* d_in, const int* in_sizes, int n_in,
                              void* d_out, int out_size, void* d_ws, size_t ws_size,
                              hipStream_t stream) {
  (void)in_sizes; (void)n_in; (void)out_size; (void)ws_size;
  const int*   ids   = (const int*)  d_in[0];
  const float* emb   = (const float*)d_in[1];
  const float* pos   = (const float*)d_in[2];
  const float* Wih_h = (const float*)d_in[3];
  const float* bih_h = (const float*)d_in[4];
  const float* Whh_h = (const float*)d_in[5];
  const float* bhh_h = (const float*)d_in[6];
  const float* Win   = (const float*)d_in[7];
  const float* bin_  = (const float*)d_in[8];
  const float* Wq    = (const float*)d_in[9];
  const float* Wk    = (const float*)d_in[10];
  const float* Wv    = (const float*)d_in[11];
  const float* Wo    = (const float*)d_in[12];
  const float* Wih_l = (const float*)d_in[13];
  const float* bih_l = (const float*)d_in[14];
  const float* Whh_l = (const float*)d_in[15];
  const float* bhh_l = (const float*)d_in[16];
  const float* ln_g  = (const float*)d_in[17];
  const float* ln_b  = (const float*)d_in[18];
  const float* Wout  = (const float*)d_in[19];
  float* out = (float*)d_out;

  char* ws = (char*)d_ws;
  size_t off = 0;
  auto alloc = [&](size_t bytes) -> char* {
    char* p = ws + off; off = (off + bytes + 255) & ~(size_t)255; return p;
  };

  // packed bf16 weights (fragment layout)
  __bf16* pWih_h[2]; __bf16* pWhh_h[2]; __bf16* pWih_l[2]; __bf16* pWhh_l[2];
  for (int l = 0; l < 2; ++l) { pWih_h[l] = (__bf16*)alloc((size_t)3072 * 1024 * 2);
                                pWhh_h[l] = (__bf16*)alloc((size_t)3072 * 1024 * 2); }
  for (int l = 0; l < 2; ++l) { pWih_l[l] = (__bf16*)alloc((size_t)3072 * 1024 * 2);
                                pWhh_l[l] = (__bf16*)alloc((size_t)3072 * 1024 * 2); }
  __bf16* pWin  = (__bf16*)alloc((size_t)1024 * 2048 * 2);
  __bf16* pWq   = (__bf16*)alloc((size_t)1024 * 1024 * 2);
  __bf16* pWk   = (__bf16*)alloc((size_t)1024 * 1024 * 2);
  __bf16* pWv   = (__bf16*)alloc((size_t)1024 * 1024 * 2);
  __bf16* pWo   = (__bf16*)alloc((size_t)1024 * 1024 * 2);
  __bf16* pWout = (__bf16*)alloc((size_t)Vv * 1024 * 2);

  // state + per-step activations
  float* hsS  = (float*)alloc((size_t)4 * 2 * 2 * 8192 * 4);   // [slot][layer][parity][8][1024]
  float* lsS  = (float*)alloc((size_t)16 * 2 * 2 * 8192 * 4);
  float* hist = (float*)alloc((size_t)Bb * 4 * 1024 * 4);
  float* Kc   = (float*)alloc((size_t)Bb * 4 * 1024 * 4);
  float* Vc   = (float*)alloc((size_t)Bb * 4 * 1024 * 4);
  float* e    = (float*)alloc((size_t)8192 * 4);
  float* inp  = (float*)alloc((size_t)8192 * 4);
  float* ctx  = (float*)alloc((size_t)8192 * 4);
  float* x2   = (float*)alloc((size_t)8192 * 4);
  float* y    = (float*)alloc((size_t)8192 * 4);

  // zero state every call (graph-capture safe)
  hipMemsetAsync(hsS,  0, (size_t)4 * 2 * 2 * 8192 * 4, stream);
  hipMemsetAsync(lsS,  0, (size_t)16 * 2 * 2 * 8192 * 4, stream);
  hipMemsetAsync(hist, 0, (size_t)Bb * 4 * 1024 * 4, stream);
  hipMemsetAsync(Kc,   0, (size_t)Bb * 4 * 1024 * 4, stream);
  hipMemsetAsync(Vc,   0, (size_t)Bb * 4 * 1024 * 4, stream);

  // one-time weight repack (deterministic; re-done each launch)
  for (int l = 0; l < 2; ++l) {
    k_pack<<<4096, 256, 0, stream>>>(Wih_h + (size_t)l * 3072 * 1024, pWih_h[l], 3072, 1024);
    k_pack<<<4096, 256, 0, stream>>>(Whh_h + (size_t)l * 3072 * 1024, pWhh_h[l], 3072, 1024);
    k_pack<<<4096, 256, 0, stream>>>(Wih_l + (size_t)l * 3072 * 1024, pWih_l[l], 3072, 1024);
    k_pack<<<4096, 256, 0, stream>>>(Whh_l + (size_t)l * 3072 * 1024, pWhh_l[l], 3072, 1024);
  }
  k_pack<<<4096, 256, 0, stream>>>(Win,  pWin,  1024, 2048);
  k_pack<<<4096, 256, 0, stream>>>(Wq,   pWq,   1024, 1024);
  k_pack<<<4096, 256, 0, stream>>>(Wk,   pWk,   1024, 1024);
  k_pack<<<4096, 256, 0, stream>>>(Wv,   pWv,   1024, 1024);
  k_pack<<<4096, 256, 0, stream>>>(Wo,   pWo,   1024, 1024);
  k_pack<<<8192, 256, 0, stream>>>(Wout, pWout, Vv,   1024);

  auto hsP = [&](int s, int l, int p) { return hsS + (((size_t)(s * 2 + l) * 2 + p) * 8192); };
  auto lsP = [&](int s, int l, int p) { return lsS + (((size_t)(s * 2 + l) * 2 + p) * 8192); };

  for (int t = 0; t < Ss; ++t) {
    int ls = t & 15;
    int hs = ls >> 2;
    bool do_high = ((ls & 3) == 0);
    int m  = t >> 4;
    int rp = (m + 1) & 1;   // read parity
    int wp = m & 1;         // write parity

    k_embed<<<32, 256, 0, stream>>>(ids, emb, pos, e, t);

    if (do_high) {
      k_gru<<<16, 128, 36864, stream>>>(e,             hsP(hs, 0, rp), bih_h,        bhh_h,
                                        pWih_h[0], pWhh_h[0], hsP(hs, 0, wp));
      k_gru<<<16, 128, 36864, stream>>>(hsP(hs, 0, wp), hsP(hs, 1, rp), bih_h + 3072, bhh_h + 3072,
                                        pWih_h[1], pWhh_h[1], hsP(hs, 1, wp));
      k_kv_hist<<<33, 128, 18432, stream>>>(hsP(hs, 1, wp), hist, Kc, Vc, pWk, pWv, hs);
    }

    // inp = gelu(concat(e, hist[:,hs]) @ Win^T + bin)
    k_linear<2048><<<16, 128, 36864, stream>>>(e, hist + (size_t)hs * 1024, 4096,
                                               pWin, bin_, nullptr, inp, 1024, 1);
    k_attn<<<8, 256, 18432, stream>>>(inp, Kc, Vc, pWq, ctx, hs);
    // x2 = inp + ctx @ Wo^T
    k_linear<1024><<<16, 128, 18432, stream>>>(ctx, nullptr, 0,
                                               pWo, nullptr, inp, x2, 1024, 2);

    k_gru<<<16, 128, 36864, stream>>>(x2,             lsP(ls, 0, rp), bih_l,        bhh_l,
                                      pWih_l[0], pWhh_l[0], lsP(ls, 0, wp));
    k_gru<<<16, 128, 36864, stream>>>(lsP(ls, 0, wp), lsP(ls, 1, rp), bih_l + 3072, bhh_l + 3072,
                                      pWih_l[1], pWhh_l[1], lsP(ls, 1, wp));

    k_ln<<<1, 256, 0, stream>>>(lsP(ls, 1, wp), ln_g, ln_b, y);
    k_logits<<<500, 128, 18432, stream>>>(y, pWout, out, t);
  }
}
